// LinformerTransformerModel_60224031425095
// MI455X (gfx1250) — matrix operand order repaired and checked by reading, not compiled, not run on hardware
//
#include <hip/hip_runtime.h>
#include <math.h>

// ---------------------------------------------------------------------------
// Model constants (Linformer: B=64, N=800, D=512, H=4, dh=128, Kp=256, FF=2048)
// ---------------------------------------------------------------------------
#define MB   64
#define MN   800
#define MD   512
#define MH   4
#define MDH  128
#define MKP  256
#define MFF  2048
#define MBN  (MB*MN)          // 51200 tokens

typedef __attribute__((ext_vector_type(16))) __bf16 v16bf;
typedef __attribute__((ext_vector_type(8)))  float  v8f;

#define DEV static __device__ __forceinline__

DEV unsigned short f2bf(float f) {
  unsigned int u = __float_as_uint(f);
  u += 0x7FFFu + ((u >> 16) & 1u);          // round-to-nearest-even
  return (unsigned short)(u >> 16);
}

// CDNA5 async global->LDS DMA (ASYNCcnt-tracked, no VGPR round-trip).
// LDS address = low 32 bits of the generic pointer (flat LDS aperture rule).
DEV void async_copy_b128(void* lds_dst, const void* gsrc) {
  unsigned lds = (unsigned)(size_t)lds_dst;
  asm volatile("global_load_async_to_lds_b128 %0, %1, off"
               :: "v"(lds), "v"(gsrc) : "memory");
}
DEV void wait_async0() {
  asm volatile("s_wait_asynccnt 0x0" ::: "memory");
}

// ---------------------------------------------------------------------------
// Generic WMMA GEMM:  C[M,N] (+epilogue) = A[M,K](bf16,row-major) * B
//   BT=1 : B stored as [N,K] row-major. Async double-buffered LDS pipeline.
//   BT=0 : B stored as [K,N] row-major -> transposed into LDS (VGPR staging).
// Workgroup: 256 thr = 8 waves; block tile 128x128; wave tile 64x32 (4x2 WMMA)
// Batched via blockIdx.z: z -> (zo=z/zdiv, zi=z%zdiv) with per-operand strides.
// LDS tile pitch = 40 halfwords (80B): 16B-aligned rows, conflict-free b128.
// ---------------------------------------------------------------------------
enum { EPI_NONE = 0, EPI_BIAS = 1, EPI_BIAS_GELU = 2, EPI_BIAS_RES = 3 };

#define TILE_HW   5120          // halfwords per 128x40 tile
#define BUF_HW    (2*TILE_HW)   // A tile + B tile

template<int EPI, int BT, bool OUT_BF16>
__global__ __launch_bounds__(256) void gemm_wmma(
    const unsigned short* __restrict__ A, int lda, long long sAo, long long sAi,
    const unsigned short* __restrict__ B, int ldb, long long sBo, long long sBi,
    void* __restrict__ Cv, int ldc, long long sCo, long long sCi,
    const float* __restrict__ bias,
    const float* __restrict__ resid, int ldr,
    int M, int N, int Kd, int zdiv)
{
  __shared__ unsigned short smem[BT ? 2 * BUF_HW : BUF_HW];

  const int t  = threadIdx.x;
  const int z  = blockIdx.z;
  const int zo = z / zdiv, zi = z % zdiv;
  A += zo * sAo + zi * sAi;
  B += zo * sBo + zi * sBi;
  const int m0 = blockIdx.y * 128;
  const int n0 = blockIdx.x * 128;

  const int wid  = t >> 5;
  const int lane = t & 31;
  const int lr   = lane & 15;
  const int lh   = lane >> 4;
  const int wm   = (wid & 1) * 64;     // wave M offset inside block tile
  const int wn   = (wid >> 1) * 32;    // wave N offset inside block tile

  const v8f vzero = {0.f,0.f,0.f,0.f,0.f,0.f,0.f,0.f};
  v8f acc[4][2];
  #pragma unroll
  for (int i = 0; i < 4; ++i)
    #pragma unroll
    for (int j = 0; j < 2; ++j) acc[i][j] = vzero;

  // fragment loads (ISA 16-bit A 16x32 / B 32x16 layouts) + 8 WMMAs
  auto compute_step = [&](const unsigned short* Asb, const unsigned short* Bsb) {
    v16bf afr[4], bfr[2];
    #pragma unroll
    for (int i = 0; i < 4; ++i) {
      int row = wm + i * 16 + lr;
      union { uint4 u[2]; v16bf v; } tmp;
      tmp.u[0] = *(const uint4*)(Asb + row * 40 + lh * 8);       // K = lh*8..+7
      tmp.u[1] = *(const uint4*)(Asb + row * 40 + 16 + lh * 8);  // K = 16+lh*8..+7
      afr[i] = tmp.v;
    }
    #pragma unroll
    for (int j = 0; j < 2; ++j) {
      int col = wn + j * 16 + lr;
      union { uint4 u[2]; v16bf v; } tmp;
      tmp.u[0] = *(const uint4*)(Bsb + col * 40 + lh * 8);       // K = lh*8..+7
      tmp.u[1] = *(const uint4*)(Bsb + col * 40 + 16 + lh * 8);  // K = 16+lh*8..+7
      bfr[j] = tmp.v;
    }
    #pragma unroll
    for (int i = 0; i < 4; ++i)
      #pragma unroll
      for (int j = 0; j < 2; ++j)
        acc[i][j] = __builtin_amdgcn_wmma_f32_16x16x32_bf16(
            false, afr[i], false, bfr[j], (short)0, acc[i][j], false, false);
  };

  if constexpr (BT == 1) {
    // ---- async double-buffered pipeline: DMA tile i+1 while WMMA on tile i --
    const int nk = Kd >> 5;
    auto issue_tile = [&](int i, unsigned short* buf) {
      const int kb = i << 5;
      unsigned short* Asb = buf;
      unsigned short* Bsb = buf + TILE_HW;
      #pragma unroll
      for (int p = 0; p < 2; ++p) {
        int r = p * 64 + (t >> 2);
        int c = (t & 3) * 8;
        long long gr = m0 + r; if (gr > (long long)M - 1) gr = M - 1;  // clamp
        async_copy_b128(Asb + r * 40 + c, A + gr * (long long)lda + kb + c);
        async_copy_b128(Bsb + r * 40 + c, B + (long long)(n0 + r) * ldb + kb + c);
      }
    };
    issue_tile(0, smem);
    for (int i = 0; i < nk; ++i) {
      wait_async0();           // my tile-i DMAs landed in LDS
      __syncthreads();         // everyone's landed; everyone done with buf^1
      if (i + 1 < nk) issue_tile(i + 1, smem + ((i + 1) & 1) * BUF_HW);
      const unsigned short* buf = smem + (i & 1) * BUF_HW;
      compute_step(buf, buf + TILE_HW);
    }
  } else {
    // ---- B stored [K,N]: transpose through LDS; batch loads before stores --
    unsigned short* Asb = smem;
    unsigned short* Bsb = smem + TILE_HW;
    for (int k0 = 0; k0 < Kd; k0 += 32) {
      __syncthreads();
      uint4 va[2], vb[2];
      #pragma unroll
      for (int p = 0; p < 2; ++p) {                 // A rows
        int r = p * 64 + (t >> 2);
        int c = (t & 3) * 8;
        long long gr = m0 + r; if (gr > (long long)M - 1) gr = M - 1;
        va[p] = *(const uint4*)(A + gr * (long long)lda + k0 + c);
      }
      #pragma unroll
      for (int p = 0; p < 2; ++p) {                 // B rows (k-major)
        int kr = p * 16 + (t >> 4);
        int c  = (t & 15) * 8;
        vb[p] = *(const uint4*)(B + (long long)(k0 + kr) * ldb + n0 + c);
      }
      #pragma unroll
      for (int p = 0; p < 2; ++p) {
        int r = p * 64 + (t >> 2);
        int c = (t & 3) * 8;
        *(uint4*)(Asb + r * 40 + c) = va[p];
      }
      #pragma unroll
      for (int p = 0; p < 2; ++p) {
        int kr = p * 16 + (t >> 4);
        int c  = (t & 15) * 8;
        unsigned short tmp[8];
        *(uint4*)tmp = vb[p];
        #pragma unroll
        for (int j = 0; j < 8; ++j) Bsb[(c + j) * 40 + kr] = tmp[j];
      }
      __syncthreads();
      compute_step(Asb, Bsb);
    }
  }

  // ---- epilogue + store (C layout: m = lh*8+reg, n = lr) ----
  float* Cf = (float*)Cv;
  unsigned short* Cb = (unsigned short*)Cv;
  const long long coff = zo * sCo + zi * sCi;
  #pragma unroll
  for (int i = 0; i < 4; ++i) {
    #pragma unroll
    for (int j = 0; j < 2; ++j) {
      int col = n0 + wn + j * 16 + lr;
      #pragma unroll
      for (int r = 0; r < 8; ++r) {
        int row = m0 + wm + i * 16 + lh * 8 + r;
        if (row < M) {
          float v = acc[i][j][r];
          if (EPI == EPI_BIAS || EPI == EPI_BIAS_GELU || EPI == EPI_BIAS_RES) v += bias[col];
          if (EPI == EPI_BIAS_GELU) v = 0.5f * v * (1.f + erff(v * 0.7071067811865475f));
          if (EPI == EPI_BIAS_RES)  v += resid[(long long)row * ldr + col];
          long long o = coff + (long long)row * ldc + col;
          if (OUT_BF16) Cb[o] = f2bf(v); else Cf[o] = v;
        }
      }
    }
  }
}

// ---------------------------------------------------------------------------
// Embedding: h = x @ emb_W + emb_b + pos     (IN_DIM=4)
// ---------------------------------------------------------------------------
__global__ __launch_bounds__(256) void embed_kernel(
    const float* __restrict__ x, const float* __restrict__ W,
    const float* __restrict__ b, const float* __restrict__ pos,
    float* __restrict__ H)
{
  long long idx = (long long)blockIdx.x * 256 + threadIdx.x;
  if (idx >= (long long)MBN * MD) return;
  int d = (int)(idx & (MD - 1));
  long long bn = idx >> 9;
  int n = (int)(bn % MN);
  const float* xr = x + bn * 4;
  float acc = b[d] + pos[(long long)n * MD + d];
  acc += xr[0] * W[d] + xr[1] * W[MD + d] + xr[2] * W[2 * MD + d] + xr[3] * W[3 * MD + d];
  H[idx] = acc;
}

// ---------------------------------------------------------------------------
// LayerNorm over D=512 -> bf16.  One wave per row (wave32: 16 elems/lane).
// ---------------------------------------------------------------------------
__global__ __launch_bounds__(256) void layernorm_bf16(
    const float* __restrict__ X, const float* __restrict__ g,
    const float* __restrict__ b, unsigned short* __restrict__ Y, int rows)
{
  int wid = threadIdx.x >> 5, lane = threadIdx.x & 31;
  long long row = (long long)blockIdx.x * 8 + wid;
  if (row >= rows) return;
  const float* xr = X + row * MD;
  float4 v[4];
  float s = 0.f, s2 = 0.f;
  #pragma unroll
  for (int i = 0; i < 4; ++i) {
    v[i] = *(const float4*)(xr + lane * 4 + i * 128);
    s  += v[i].x + v[i].y + v[i].z + v[i].w;
    s2 += v[i].x * v[i].x + v[i].y * v[i].y + v[i].z * v[i].z + v[i].w * v[i].w;
  }
  #pragma unroll
  for (int off = 16; off > 0; off >>= 1) {
    s  += __shfl_xor(s,  off, 32);
    s2 += __shfl_xor(s2, off, 32);
  }
  float mu   = s * (1.f / MD);
  float var  = s2 * (1.f / MD) - mu * mu;
  float rstd = rsqrtf(var + 1e-5f);
  #pragma unroll
  for (int i = 0; i < 4; ++i) {
    int c = lane * 4 + i * 128;
    float4 gg = *(const float4*)(g + c);
    float4 bb = *(const float4*)(b + c);
    unsigned short* yr = Y + row * MD + c;
    yr[0] = f2bf((v[i].x - mu) * rstd * gg.x + bb.x);
    yr[1] = f2bf((v[i].y - mu) * rstd * gg.y + bb.y);
    yr[2] = f2bf((v[i].z - mu) * rstd * gg.z + bb.z);
    yr[3] = f2bf((v[i].w - mu) * rstd * gg.w + bb.w);
  }
}

// ---------------------------------------------------------------------------
// Row softmax over 256 logits (scale folded in) -> bf16.  One wave per row.
// ---------------------------------------------------------------------------
__global__ __launch_bounds__(256) void softmax256_bf16(
    const float* __restrict__ D, unsigned short* __restrict__ O,
    long long rows, float scale)
{
  int wid = threadIdx.x >> 5, lane = threadIdx.x & 31;
  long long row = (long long)blockIdx.x * 8 + wid;
  if (row >= rows) return;
  const float* dr = D + row * 256;
  float4 a = *(const float4*)(dr + lane * 8);
  float4 c = *(const float4*)(dr + lane * 8 + 4);
  float e[8] = { a.x*scale, a.y*scale, a.z*scale, a.w*scale,
                 c.x*scale, c.y*scale, c.z*scale, c.w*scale };
  float mx = e[0];
  #pragma unroll
  for (int i = 1; i < 8; ++i) mx = fmaxf(mx, e[i]);
  #pragma unroll
  for (int off = 16; off > 0; off >>= 1) mx = fmaxf(mx, __shfl_xor(mx, off, 32));
  float sum = 0.f;
  #pragma unroll
  for (int i = 0; i < 8; ++i) { e[i] = __expf(e[i] - mx); sum += e[i]; }
  #pragma unroll
  for (int off = 16; off > 0; off >>= 1) sum += __shfl_xor(sum, off, 32);
  float inv = 1.f / sum;
  unsigned short* orow = O + row * 256 + lane * 8;
  #pragma unroll
  for (int i = 0; i < 8; ++i) orow[i] = f2bf(e[i] * inv);
}

// ---------------------------------------------------------------------------
// Weight prep: fp32 [R,C] -> bf16 transposed [C,R]
// ---------------------------------------------------------------------------
__global__ __launch_bounds__(256) void transpose_bf16(
    const float* __restrict__ S, unsigned short* __restrict__ D, int R, int C)
{
  long long idx = (long long)blockIdx.x * 256 + threadIdx.x;
  if (idx >= (long long)R * C) return;
  int c = (int)(idx % C);
  long long r = idx / C;
  D[(long long)c * R + r] = f2bf(S[idx]);
}

// ---------------------------------------------------------------------------
// out[b] = mean_n(h[b]) @ fc_W + fc_b   (D=512 -> 2 classes)
// ---------------------------------------------------------------------------
__global__ __launch_bounds__(256) void mean_fc(
    const float* __restrict__ H, const float* __restrict__ W,
    const float* __restrict__ b, float* __restrict__ out)
{
  int bi = blockIdx.x;
  int t  = threadIdx.x;
  float p0 = 0.f, p1 = 0.f;
  for (int d = t; d < MD; d += 256) {
    const float* col = H + (long long)bi * MN * MD + d;
    float s = 0.f;
    for (int n = 0; n < MN; ++n) s += col[(long long)n * MD];
    s *= (1.f / MN);
    p0 += s * W[d * 2 + 0];
    p1 += s * W[d * 2 + 1];
  }
  __shared__ float r0[256], r1[256];
  r0[t] = p0; r1[t] = p1;
  __syncthreads();
  for (int off = 128; off > 0; off >>= 1) {
    if (t < off) { r0[t] += r0[t + off]; r1[t] += r1[t + off]; }
    __syncthreads();
  }
  if (t == 0) {
    out[bi * 2 + 0] = r0[0] + b[0];
    out[bi * 2 + 1] = r1[0] + b[1];
  }
}

// ---------------------------------------------------------------------------
// Host orchestration
// ---------------------------------------------------------------------------
extern "C" void kernel_launch(void* const* d_in, const int* in_sizes, int n_in,
                              void* d_out, int out_size, void* d_ws, size_t ws_size,
                              hipStream_t stream)
{
  (void)in_sizes; (void)n_in; (void)out_size; (void)ws_size;
  const float* x    = (const float*)d_in[0];
  const float* embW = (const float*)d_in[1];
  const float* embB = (const float*)d_in[2];
  const float* pos  = (const float*)d_in[3];
  const float* fcW  = (const float*)d_in[4];
  const float* fcB  = (const float*)d_in[5];

  struct LayerP {
    const float *ln1g, *ln1b, *Wq, *Wk, *Wv, *pk, *pv, *Wo, *bo, *ln2g, *ln2b, *W1, *b1, *W2, *b2;
  } L[2];
  for (int l = 0; l < 2; ++l) {
    const float** f = (const float**)&L[l];
    for (int i = 0; i < 15; ++i) f[i] = (const float*)d_in[6 + l * 15 + i];
  }

  // workspace carve-out
  char* p = (char*)d_ws;
  auto alloc = [&](size_t bytes) -> void* {
    void* r = (void*)p;
    p += (bytes + 255) & ~(size_t)255;
    return r;
  };
  float*          HB    = (float*)alloc((size_t)MBN * MD * 4);
  float*          H2    = (float*)alloc((size_t)MBN * MD * 4);
  unsigned short* XN    = (unsigned short*)alloc((size_t)MBN * MD * 2);
  unsigned short* Q     = (unsigned short*)alloc((size_t)MBN * MD * 2);
  unsigned short* K0    = (unsigned short*)alloc((size_t)MBN * MD * 2);   // aliased by ATTNO
  unsigned short* V0    = (unsigned short*)alloc((size_t)MBN * MD * 2);
  unsigned short* KP    = (unsigned short*)alloc((size_t)MB * MKP * MD * 2);
  unsigned short* VP    = (unsigned short*)alloc((size_t)MB * MKP * MD * 2);
  float*          DOTS  = (float*)alloc((size_t)MB * MH * MN * MKP * 4);  // aliased by HMID
  unsigned short* ATTN  = (unsigned short*)alloc((size_t)MB * MH * MN * MKP * 2);
  unsigned short* ATTNO = K0;              // lifetime-disjoint alias
  unsigned short* HMID  = (unsigned short*)DOTS;  // lifetime-disjoint alias

  unsigned short *WqT[2], *WkT[2], *WvT[2], *WoT[2], *W1T[2], *W2T[2], *PKT[2], *PVT[2];
  for (int l = 0; l < 2; ++l) {
    WqT[l] = (unsigned short*)alloc((size_t)MD * MD * 2);
    WkT[l] = (unsigned short*)alloc((size_t)MD * MD * 2);
    WvT[l] = (unsigned short*)alloc((size_t)MD * MD * 2);
    WoT[l] = (unsigned short*)alloc((size_t)MD * MD * 2);
    W1T[l] = (unsigned short*)alloc((size_t)MD * MFF * 2);
    W2T[l] = (unsigned short*)alloc((size_t)MFF * MD * 2);
    PKT[l] = (unsigned short*)alloc((size_t)MN * MKP * 2);
    PVT[l] = (unsigned short*)alloc((size_t)MN * MKP * 2);
  }

  auto blocks = [](long long n) { return (unsigned)((n + 255) / 256); };

  // --- weight prep (per launch; deterministic) ---
  for (int l = 0; l < 2; ++l) {
    transpose_bf16<<<blocks((long long)MD * MD), 256, 0, stream>>>(L[l].Wq, WqT[l], MD, MD);
    transpose_bf16<<<blocks((long long)MD * MD), 256, 0, stream>>>(L[l].Wk, WkT[l], MD, MD);
    transpose_bf16<<<blocks((long long)MD * MD), 256, 0, stream>>>(L[l].Wv, WvT[l], MD, MD);
    transpose_bf16<<<blocks((long long)MD * MD), 256, 0, stream>>>(L[l].Wo, WoT[l], MD, MD);
    transpose_bf16<<<blocks((long long)MD * MFF), 256, 0, stream>>>(L[l].W1, W1T[l], MD, MFF);
    transpose_bf16<<<blocks((long long)MFF * MD), 256, 0, stream>>>(L[l].W2, W2T[l], MFF, MD);
    transpose_bf16<<<blocks((long long)MN * MKP), 256, 0, stream>>>(L[l].pk, PKT[l], MN, MKP);
    transpose_bf16<<<blocks((long long)MN * MKP), 256, 0, stream>>>(L[l].pv, PVT[l], MN, MKP);
  }

  // --- embedding ---
  embed_kernel<<<blocks((long long)MBN * MD), 256, 0, stream>>>(x, embW, embB, pos, HB);

  const long long sQ  = (long long)MN * MD;     // per-batch stride in Q/K0/V0/ATTNO
  const long long sKV = (long long)MKP * MD;    // per-batch stride in KP/VP
  const long long sDh = (long long)MN * MKP;    // per-head stride in DOTS/ATTN

  for (int l = 0; l < 2; ++l) {
    // LN1 -> bf16
    layernorm_bf16<<<MBN / 8, 256, 0, stream>>>(HB, L[l].ln1g, L[l].ln1b, XN, MBN);

    // Q/K0/V0 = XN @ W{q,k,v}      [51200 x 512 x 512]
    dim3 gq(MD / 128, MBN / 128, 1);
    gemm_wmma<EPI_NONE, 1, true><<<gq, 256, 0, stream>>>(
        XN, MD, 0, 0, WqT[l], MD, 0, 0, Q, MD, 0, 0, nullptr, nullptr, 0, MBN, MD, MD, 1);
    gemm_wmma<EPI_NONE, 1, true><<<gq, 256, 0, stream>>>(
        XN, MD, 0, 0, WkT[l], MD, 0, 0, K0, MD, 0, 0, nullptr, nullptr, 0, MBN, MD, MD, 1);
    gemm_wmma<EPI_NONE, 1, true><<<gq, 256, 0, stream>>>(
        XN, MD, 0, 0, WvT[l], MD, 0, 0, V0, MD, 0, 0, nullptr, nullptr, 0, MBN, MD, MD, 1);

    // KP[b] = proj_k^T @ K0[b]     [256 x 800 x 512], batched over b, B is [K,N]
    dim3 gp(MD / 128, MKP / 128, MB);
    gemm_wmma<EPI_NONE, 0, true><<<gp, 256, 0, stream>>>(
        PKT[l], MN, 0, 0, K0, MD, sQ, 0, KP, MD, sKV, 0, nullptr, nullptr, 0, MKP, MD, MN, 1);
    gemm_wmma<EPI_NONE, 0, true><<<gp, 256, 0, stream>>>(
        PVT[l], MN, 0, 0, V0, MD, sQ, 0, VP, MD, sKV, 0, nullptr, nullptr, 0, MKP, MD, MN, 1);

    // DOTS[b,h] = Q[b,:,h] @ KP[b,:,h]^T   [800 x 128 x 256]; KP is already [N,K]
    dim3 gd(MKP / 128, (MN + 127) / 128, MB * MH);
    gemm_wmma<EPI_NONE, 1, false><<<gd, 256, 0, stream>>>(
        Q, MD, sQ, MDH, KP, MD, sKV, MDH, DOTS, MKP, (long long)MH * sDh, sDh,
        nullptr, nullptr, 0, MN, MKP, MDH, MH);

    // softmax (scale = dh^-0.5)
    long long srows = (long long)MB * MH * MN;
    softmax256_bf16<<<(unsigned)((srows + 7) / 8), 256, 0, stream>>>(
        DOTS, ATTN, srows, 0.08838834764831845f);

    // ATTNO[b,:,h] = ATTN[b,h] @ VP[b,:,h]   [800 x 256 x 128]; B is [K,N]
    dim3 ga(MDH / 128, (MN + 127) / 128, MB * MH);
    gemm_wmma<EPI_NONE, 0, true><<<ga, 256, 0, stream>>>(
        ATTN, MKP, (long long)MH * sDh, sDh, VP, MD, sKV, MDH, ATTNO, MD, sQ, MDH,
        nullptr, nullptr, 0, MN, MDH, MKP, MH);

    // H2 = HB + ATTNO @ Wo + bo
    gemm_wmma<EPI_BIAS_RES, 1, false><<<gq, 256, 0, stream>>>(
        ATTNO, MD, 0, 0, WoT[l], MD, 0, 0, H2, MD, 0, 0, L[l].bo, HB, MD, MBN, MD, MD, 1);

    // LN2 -> bf16
    layernorm_bf16<<<MBN / 8, 256, 0, stream>>>(H2, L[l].ln2g, L[l].ln2b, XN, MBN);

    // HMID = gelu(XN @ W1 + b1)    [51200 x 512 x 2048]
    dim3 g1(MFF / 128, MBN / 128, 1);
    gemm_wmma<EPI_BIAS_GELU, 1, true><<<g1, 256, 0, stream>>>(
        XN, MD, 0, 0, W1T[l], MD, 0, 0, HMID, MFF, 0, 0, L[l].b1, nullptr, 0, MBN, MFF, MD, 1);

    // HB = H2 + HMID @ W2 + b2     [51200 x 2048 x 512]
    gemm_wmma<EPI_BIAS_RES, 1, false><<<gq, 256, 0, stream>>>(
        HMID, MFF, 0, 0, W2T[l], MFF, 0, 0, HB, MD, 0, 0, L[l].b2, H2, MD, MBN, MD, MFF, 1);
  }

  // --- mean over sequence + classifier ---
  mean_fc<<<MB, 256, 0, stream>>>(HB, fcW, fcB, (float*)d_out);
}
